// SpatialGAT_17635135717842
// MI455X (gfx1250) — matrix operand order repaired and checked by reading, not compiled, not run on hardware
//
#include <hip/hip_runtime.h>

#define NN   50000
#define EE   800000
#define DINC 256
#define HIDC 64
#define NHEADS 4
#define EDIMC 16

typedef __attribute__((ext_vector_type(16))) __bf16 v16bf;
typedef __attribute__((ext_vector_type(8)))  __bf16 v8bf;
typedef __attribute__((ext_vector_type(8)))  float  v8f;

__device__ __forceinline__ unsigned short f2bf(float f) {
    unsigned u = __float_as_uint(f);
    unsigned r = 0x7FFFu + ((u >> 16) & 1u);   // round-to-nearest-even
    return (unsigned short)((u + r) >> 16);
}
// monotone float<->uint encoding so unsigned atomicMax == float max
__device__ __forceinline__ unsigned enc_f32(float f) {
    unsigned u = __float_as_uint(f);
    return (u & 0x80000000u) ? ~u : (u | 0x80000000u);
}
__device__ __forceinline__ float dec_f32(unsigned u) {
    unsigned b = (u & 0x80000000u) ? (u ^ 0x80000000u) : ~u;
    return __uint_as_float(b);
}

// ---------------- small utility kernels ----------------
__global__ void k_fill_u32(unsigned* p, unsigned v, int n) {
    int i = blockIdx.x * blockDim.x + threadIdx.x;
    if (i < n) p[i] = v;
}
__global__ void k_fill_f32(float* p, float v, int n) {
    int i = blockIdx.x * blockDim.x + threadIdx.x;
    if (i < n) p[i] = v;
}
__global__ void k_init_bias(float* out, const float* b, int total, int ch) {
    int i = blockIdx.x * blockDim.x + threadIdx.x;
    if (i < total) out[i] = b[i % ch];
}
__global__ void k_f32_to_bf16(const float* s, unsigned short* d, int n) {
    int i = blockIdx.x * blockDim.x + threadIdx.x;
    if (i < n) d[i] = f2bf(s[i]);
}
__global__ void k_relu_to_bf16(const float* s, unsigned short* d, int n) {
    int i = blockIdx.x * blockDim.x + threadIdx.x;
    if (i < n) d[i] = f2bf(fmaxf(s[i], 0.f));
}
// W [K,N] row-major (f32) -> Wt [N,K] row-major (bf16)
__global__ void k_transpose_bf16(const float* W, unsigned short* Wt, int K, int Ncol) {
    int i = blockIdx.x * blockDim.x + threadIdx.x;
    if (i < K * Ncol) {
        int k = i / Ncol, n = i % Ncol;
        Wt[(size_t)n * K + k] = f2bf(W[i]);
    }
}

// P1[k,h] = sum_c We1[k, h*64+c] * a1_edge[h,c];  P2[k] = sum_c We2[k,c]*a2_edge[c]
__global__ void k_precompute_P(const float* We1, const float* a1e,
                               const float* We2, const float* a2e,
                               float* P1, float* P2) {
    int t = threadIdx.x;
    if (t < EDIMC * NHEADS) {
        int k = t & 15, h = t >> 4;
        float s = 0.f;
        for (int c = 0; c < HIDC; ++c)
            s += We1[k * (NHEADS * HIDC) + h * HIDC + c] * a1e[h * HIDC + c];
        P1[k * NHEADS + h] = s;
    } else if (t < EDIMC * NHEADS + EDIMC) {
        int k = t - EDIMC * NHEADS;
        float s = 0.f;
        for (int c = 0; c < HIDC; ++c)
            s += We2[k * HIDC + c] * a2e[c];
        P2[k] = s;
    }
}

// per edge: e_k = relu(edge_attr @ W_sp + b_sp); ae1[e,h] = e@P1; ae2[e] = e@P2
__global__ void k_edge_prep(const float* eattr, const float* Wsp, const float* bsp,
                            const float* P1, const float* P2, float* ae1, float* ae2) {
    int e = blockIdx.x * blockDim.x + threadIdx.x;
    if (e >= EE) return;
    float u0 = eattr[2 * e], u1 = eattr[2 * e + 1];
    float acc[NHEADS] = {0.f, 0.f, 0.f, 0.f};
    float acc2 = 0.f;
    #pragma unroll
    for (int k = 0; k < EDIMC; ++k) {
        float ek = fmaxf(u0 * Wsp[k] + u1 * Wsp[EDIMC + k] + bsp[k], 0.f);
        acc2 += ek * P2[k];
        #pragma unroll
        for (int h = 0; h < NHEADS; ++h) acc[h] += ek * P1[k * NHEADS + h];
    }
    #pragma unroll
    for (int h = 0; h < NHEADS; ++h) ae1[e * NHEADS + h] = acc[h];
    ae2[e] = acc2;
}

// ---------------- WMMA GEMM: C[M,N] = A[M,K](bf16) x Bt[N,K](bf16) ----------------
// one wave computes a 16x64 strip of C; K in steps of 32 via v_wmma_f32_16x16x32_bf16
__global__ void __launch_bounds__(256)
k_gemm_wmma(const unsigned short* __restrict__ A, const unsigned short* __restrict__ Bt,
            float* __restrict__ C, int M, int Ncol, int K) {
    const int lane = threadIdx.x & 31;
    const int wid  = threadIdx.x >> 5;
    const int gw   = blockIdx.x * 8 + wid;
    const int nGroups = Ncol >> 6;           // N in groups of 64
    const int nTilesM = M >> 4;
    if (gw >= nTilesM * nGroups) return;     // wave-uniform: EXEC stays all-ones
    const int mt   = gw / nGroups;
    const int ng   = gw % nGroups;
    const int half = lane >> 4;
    const int l16  = lane & 15;

    v8f acc[4];
    #pragma unroll
    for (int s = 0; s < 4; ++s) { v8f z = {}; acc[s] = z; }

    const unsigned short* arow = A + (size_t)(mt * 16 + l16) * K;
    for (int k0 = 0; k0 < K; k0 += 32) {
        // A fragment (16x32 bf16): lane=row l16; K = {k0+8*half..+7} and {k0+16+8*half..+7}
        v16bf af;
        {
            const v8bf alo = *(const v8bf*)(arow + k0 + half * 8);
            const v8bf ahi = *(const v8bf*)(arow + k0 + 16 + half * 8);
            #pragma unroll
            for (int i = 0; i < 8; ++i) { af[i] = alo[i]; af[i + 8] = ahi[i]; }
        }
        #pragma unroll
        for (int s = 0; s < 4; ++s) {
            // B fragment (32x16 bf16): lane=col l16; K = {k0+8*half..+7} and {k0+16+8*half..+7}
            const int col = ng * 64 + s * 16 + l16;
            const unsigned short* bp = Bt + (size_t)col * K + k0 + half * 8;
            v16bf bf;
            const v8bf b0 = *(const v8bf*)bp;
            const v8bf b1 = *(const v8bf*)(bp + 16);
            #pragma unroll
            for (int i = 0; i < 8; ++i) { bf[i] = b0[i]; bf[i + 8] = b1[i]; }
            acc[s] = __builtin_amdgcn_wmma_f32_16x16x32_bf16(
                false, af, false, bf, (short)0, acc[s], false, false);
        }
    }
    // C/D layout: VGPR r -> row = mt*16 + half*8 + r, col = ng*64 + s*16 + l16
    #pragma unroll
    for (int s = 0; s < 4; ++s) {
        #pragma unroll
        for (int r = 0; r < 8; ++r) {
            C[(size_t)(mt * 16 + half * 8 + r) * Ncol + ng * 64 + s * 16 + l16] = acc[s][r];
        }
    }
}

// a_s[n,h] = <xs[n,h,:], a_src[h]>, a_d likewise
__global__ void k_node_scores(const float* __restrict__ xs, const float* __restrict__ asrc,
                              const float* __restrict__ adst, float* as_o, float* ad_o,
                              int nodes, int heads) {
    int t = blockIdx.x * blockDim.x + threadIdx.x;
    if (t >= nodes * heads) return;
    int n = t / heads, h = t % heads;
    const float* xp = xs + (size_t)n * heads * HIDC + h * HIDC;
    const float* sp = asrc + h * HIDC;
    const float* dp = adst + h * HIDC;
    float s = 0.f, d = 0.f;
    #pragma unroll 8
    for (int c = 0; c < HIDC; ++c) { float v = xp[c]; s += v * sp[c]; d += v * dp[c]; }
    as_o[t] = s; ad_o[t] = d;
}

__global__ void k_edge_max(const int* src, const int* dst, const float* as,
                           const float* ad, const float* ae, unsigned* menc, int heads) {
    int e = blockIdx.x * blockDim.x + threadIdx.x;
    if (e >= EE) return;
    int s = src[e], d = dst[e];
    for (int h = 0; h < heads; ++h) {
        float a = as[s * heads + h] + ad[d * heads + h] + ae[e * heads + h];
        a = (a > 0.f) ? a : 0.2f * a;                // leaky_relu(0.2)
        atomicMax(&menc[d * heads + h], enc_f32(a));
    }
}

__global__ void k_edge_exp(const int* src, const int* dst, const float* as,
                           const float* ad, const float* ae, const unsigned* menc,
                           float* ex, float* den, int heads) {
    int e = blockIdx.x * blockDim.x + threadIdx.x;
    if (e >= EE) return;
    int s = src[e], d = dst[e];
    for (int h = 0; h < heads; ++h) {
        float a = as[s * heads + h] + ad[d * heads + h] + ae[e * heads + h];
        a = (a > 0.f) ? a : 0.2f * a;
        float x = __expf(a - dec_f32(menc[d * heads + h]));
        ex[e * heads + h] = x;
        atomicAdd(&den[d * heads + h], x);
    }
}

// one wave per edge: out[dst, c] += xs[src, c] * ex[e,h]/(den[dst,h]+1e-16)
__global__ void __launch_bounds__(256)
k_edge_aggregate(const int* src, const int* dst, const float* __restrict__ xs,
                 const float* __restrict__ ex, const float* __restrict__ den,
                 float* out, int heads) {
    const int lane = threadIdx.x & 31;
    const int wid  = threadIdx.x >> 5;
    const int e    = blockIdx.x * 8 + wid;
    if (e >= EE) return;
    const int s = src[e], d = dst[e];
    if (heads == 4) {                                  // 256 channels, 8 per lane
        const float4* xv = (const float4*)(xs + (size_t)s * 256 + lane * 8);
        float4 v0 = xv[0], v1 = xv[1];
        int h = lane >> 3;
        float w = ex[e * 4 + h] / (den[d * 4 + h] + 1e-16f);
        float* op = out + (size_t)d * 256 + lane * 8;
        atomicAdd(op + 0, v0.x * w); atomicAdd(op + 1, v0.y * w);
        atomicAdd(op + 2, v0.z * w); atomicAdd(op + 3, v0.w * w);
        atomicAdd(op + 4, v1.x * w); atomicAdd(op + 5, v1.y * w);
        atomicAdd(op + 6, v1.z * w); atomicAdd(op + 7, v1.w * w);
    } else {                                           // 64 channels, 2 per lane
        float2 v = *(const float2*)(xs + (size_t)s * 64 + lane * 2);
        float w = ex[e] / (den[d] + 1e-16f);
        float* op = out + (size_t)d * 64 + lane * 2;
        atomicAdd(op + 0, v.x * w); atomicAdd(op + 1, v.y * w);
    }
}

extern "C" void kernel_launch(void* const* d_in, const int* in_sizes, int n_in,
                              void* d_out, int out_size, void* d_ws, size_t ws_size,
                              hipStream_t stream) {
    (void)in_sizes; (void)n_in; (void)out_size; (void)ws_size;
    const float* x    = (const float*)d_in[0];
    const int*   eidx = (const int*)d_in[1];
    const float* eatt = (const float*)d_in[2];
    const float* Wsp  = (const float*)d_in[3];
    const float* bsp  = (const float*)d_in[4];
    const float* W1   = (const float*)d_in[5];
    const float* We1  = (const float*)d_in[6];
    const float* a1s  = (const float*)d_in[7];
    const float* a1d  = (const float*)d_in[8];
    const float* a1e  = (const float*)d_in[9];
    const float* b1   = (const float*)d_in[10];
    const float* W2   = (const float*)d_in[11];
    const float* We2  = (const float*)d_in[12];
    const float* a2s  = (const float*)d_in[13];
    const float* a2d  = (const float*)d_in[14];
    const float* a2e  = (const float*)d_in[15];
    const float* b2   = (const float*)d_in[16];
    const int* src = eidx;
    const int* dst = eidx + EE;
    float* out = (float*)d_out;

    // workspace carve-out (aligned 256B); layer-2 arrays overlay dead layer-1 arrays
    char* w = (char*)d_ws;
    size_t off = 0;
    auto take = [&](size_t bytes) -> char* {
        char* p = w + off; off = (off + bytes + 255) & ~(size_t)255; return p;
    };
    float* xs1          = (float*)take((size_t)NN * 256 * 4);
    float* out1         = (float*)take((size_t)NN * 256 * 4);
    unsigned short* xbf = (unsigned short*)take((size_t)NN * 256 * 2); // also h_bf16
    unsigned short* w1t = (unsigned short*)take(256 * 256 * 2);
    unsigned short* w2t = (unsigned short*)take(64 * 256 * 2);
    float* P1           = (float*)take(EDIMC * NHEADS * 4);
    float* P2           = (float*)take(EDIMC * 4);
    float* as1          = (float*)take((size_t)NN * 4 * 4);
    float* ad1          = (float*)take((size_t)NN * 4 * 4);
    unsigned* m1        = (unsigned*)take((size_t)NN * 4 * 4);
    float* den1         = (float*)take((size_t)NN * 4 * 4);
    float* ae1          = (float*)take((size_t)EE * 4 * 4); // reused as xs2
    float* ex1          = (float*)take((size_t)EE * 4 * 4); // reused as ex2
    float* ae2          = (float*)take((size_t)EE * 4);
    float* xs2 = ae1; float* ex2 = ex1;
    float* as2 = as1; float* ad2 = ad1;
    unsigned* m2 = m1; float* den2 = den1;

    const int B = 256;
    const unsigned NEG_INF_ENC = 0x007FFFFFu; // enc(-inf)

    // preprocessing
    k_f32_to_bf16<<<(NN * 256 + B - 1) / B, B, 0, stream>>>(x, xbf, NN * 256);
    k_transpose_bf16<<<(256 * 256 + B - 1) / B, B, 0, stream>>>(W1, w1t, 256, 256);
    k_transpose_bf16<<<(256 * 64 + B - 1) / B, B, 0, stream>>>(W2, w2t, 256, 64);
    k_precompute_P<<<1, 256, 0, stream>>>(We1, a1e, We2, a2e, P1, P2);
    k_edge_prep<<<(EE + B - 1) / B, B, 0, stream>>>(eatt, Wsp, bsp, P1, P2, ae1, ae2);

    // ---- layer 1 (heads=4) ----
    k_fill_u32<<<(NN * 4 + B - 1) / B, B, 0, stream>>>(m1, NEG_INF_ENC, NN * 4);
    k_fill_f32<<<(NN * 4 + B - 1) / B, B, 0, stream>>>(den1, 0.f, NN * 4);
    k_init_bias<<<(NN * 256 + B - 1) / B, B, 0, stream>>>(out1, b1, NN * 256, 256);
    {   // xs1 = x @ W1 via WMMA
        int waves = (NN / 16) * (256 / 64);
        k_gemm_wmma<<<(waves + 7) / 8, 256, 0, stream>>>(xbf, w1t, xs1, NN, 256, 256);
    }
    k_node_scores<<<(NN * 4 + B - 1) / B, B, 0, stream>>>(xs1, a1s, a1d, as1, ad1, NN, 4);
    k_edge_max<<<(EE + B - 1) / B, B, 0, stream>>>(src, dst, as1, ad1, ae1, m1, 4);
    k_edge_exp<<<(EE + B - 1) / B, B, 0, stream>>>(src, dst, as1, ad1, ae1, m1, ex1, den1, 4);
    k_edge_aggregate<<<(EE + 7) / 8, 256, 0, stream>>>(src, dst, xs1, ex1, den1, out1, 4);

    // h = relu(out1) -> bf16 (reuse xbf)
    k_relu_to_bf16<<<(NN * 256 + B - 1) / B, B, 0, stream>>>(out1, xbf, NN * 256);

    // ---- layer 2 (heads=1) ----
    k_fill_u32<<<(NN + B - 1) / B, B, 0, stream>>>(m2, NEG_INF_ENC, NN);
    k_fill_f32<<<(NN + B - 1) / B, B, 0, stream>>>(den2, 0.f, NN);
    k_init_bias<<<(NN * 64 + B - 1) / B, B, 0, stream>>>(out, b2, NN * 64, 64);
    {   // xs2 = h @ W2 via WMMA
        int waves = NN / 16;
        k_gemm_wmma<<<(waves + 7) / 8, 256, 0, stream>>>(xbf, w2t, xs2, NN, 64, 256);
    }
    k_node_scores<<<(NN + B - 1) / B, B, 0, stream>>>(xs2, a2s, a2d, as2, ad2, NN, 1);
    k_edge_max<<<(EE + B - 1) / B, B, 0, stream>>>(src, dst, as2, ad2, ae2, m2, 1);
    k_edge_exp<<<(EE + B - 1) / B, B, 0, stream>>>(src, dst, as2, ad2, ae2, m2, ex2, den2, 1);
    k_edge_aggregate<<<(EE + 7) / 8, 256, 0, stream>>>(src, dst, xs2, ex2, den2, out, 1);
}
